// Attn_79491254714864
// MI455X (gfx1250) — compile-verified, matrix operand order checked by reading, not run on hardware
//
#include <hip/hip_runtime.h>
#include <math.h>

typedef __attribute__((ext_vector_type(16))) _Float16 v16h;
typedef __attribute__((ext_vector_type(8)))  float    v8f;

#define BB 16
#define SS 2048
#define DD 1024

// ---------------- kernel 0: zero scores scratch ----------------
__global__ void k_zero(float* __restrict__ p, int n) {
    int i = blockIdx.x * blockDim.x + threadIdx.x;
    if (i < n) p[i] = 0.0f;
}

// ---------------- kernel 1: hb[b][e] = hidden[b,:] . Wh[e,:] + bias[e] ----------------
// Wh[e,d] = W[e*2D + d]
__global__ __launch_bounds__(256) void k_hidproj(const float* __restrict__ hidden,
                                                 const float* __restrict__ W,
                                                 const float* __restrict__ bias,
                                                 float* __restrict__ hb) {
    int e = blockIdx.x * 256 + threadIdx.x;   // grid.x = DD/256
    int b = blockIdx.y;
    const float* h = hidden + b * DD;
    const float* w = W + (size_t)e * (2 * DD);
    float acc = 0.0f;
    #pragma unroll 4
    for (int d = 0; d < DD; d += 4) {
        float4 hv = *(const float4*)(h + d);
        float4 wv = *(const float4*)(w + d);
        acc += hv.x * wv.x + hv.y * wv.y + hv.z * wv.z + hv.w * wv.w;
    }
    hb[b * DD + e] = acc + bias[e];
}

// helper: pack 8 consecutive f32 (two float4) into v16h[base..base+7]
__device__ __forceinline__ void cvt8(v16h& dst, int base, float4 x, float4 y) {
    dst[base + 0] = (_Float16)x.x; dst[base + 1] = (_Float16)x.y;
    dst[base + 2] = (_Float16)x.z; dst[base + 3] = (_Float16)x.w;
    dst[base + 4] = (_Float16)y.x; dst[base + 5] = (_Float16)y.y;
    dst[base + 6] = (_Float16)y.z; dst[base + 7] = (_Float16)y.w;
}

__device__ __forceinline__ void loadfrag(v16h& f, const float* p, int kk0, int kk1) {
    float4 x0 = *(const float4*)(p + kk0);
    float4 x1 = *(const float4*)(p + kk0 + 4);
    float4 x2 = *(const float4*)(p + kk1);
    float4 x3 = *(const float4*)(p + kk1 + 4);
    cvt8(f, 0, x0, x1);
    cvt8(f, 8, x2, x3);
}

// ---------------- kernel 2: fused enc@We^T -> tanh -> dot(v) -> partial scores ----------------
// grid: (DD/512, SS/32, BB), block 256 = 8 waves.
// Each wave owns a 32(s) x 64(e) tile: 2 A fragments x 4 B fragments -> 8 WMMAs
// per K-step (each A reused 4x, each B reused 2x).
// Block covers e-range of 512; per-row partial scores are reduced across the
// 8 waves through LDS, then 1 atomic per row per block (2 atomics/score total).
__global__ __launch_bounds__(256) void k_energy(const float* __restrict__ enc,
                                                const float* __restrict__ W,
                                                const float* __restrict__ hb,
                                                const float* __restrict__ vvec,
                                                float* __restrict__ scores) {
    __shared__ float red[8][32];

    const int lane = threadIdx.x & 31;
    const int wave = threadIdx.x >> 5;
    const int half = lane >> 4;         // 0 or 1
    const int row  = lane & 15;         // M-row for A, N-row (=e row) for B, N-col for C
    const int b    = blockIdx.z;
    const int s0   = blockIdx.y * 32;
    const int e0   = (blockIdx.x * 8 + wave) * 64;   // wave's 64-wide e range

    // A: enc rows (M = s), row-major contiguous in K(=d); two 16-row M tiles
    const float* aRow0 = enc + ((size_t)b * SS + (size_t)(s0 + row)) * DD;
    const float* aRow1 = aRow0 + (size_t)16 * DD;
    // B: We rows (N = e): We[e,d] = W[e*2D + D + d]; tile t at constant offset t*16*2*DD
    const float* bBase = W + (size_t)(e0 + row) * (2 * DD) + DD;

    const int h8 = half * 8;
    v8f acc[2][4] = {};

    for (int k = 0; k < DD; k += 32) {
        const int kk0 = k + h8;
        const int kk1 = k + 16 + h8;

        v16h af0, af1;
        loadfrag(af0, aRow0, kk0, kk1);
        loadfrag(af1, aRow1, kk0, kk1);

        if (k + 64 < DD) {
            __builtin_prefetch(aRow0 + kk0 + 64, 0, 1);
            __builtin_prefetch(aRow1 + kk0 + 64, 0, 1);
            __builtin_prefetch(bBase + kk0 + 64, 0, 1);
        }

        #pragma unroll
        for (int t = 0; t < 4; ++t) {
            v16h bf;
            loadfrag(bf, bBase + (size_t)t * (16 * 2 * DD), kk0, kk1);
            acc[0][t] = __builtin_amdgcn_wmma_f32_16x16x32_f16(
                false, af0, false, bf, (short)0, acc[0][t], false, false);
            acc[1][t] = __builtin_amdgcn_wmma_f32_16x16x32_f16(
                false, af1, false, bf, (short)0, acc[1][t], false, false);
        }
    }

    // Epilogue: energy = tanh(acc + hb[b][e]); row-sum of energy*v[e] over the 64 e's.
    float rowp[2][8];
    #pragma unroll
    for (int m = 0; m < 2; ++m)
        #pragma unroll
        for (int r = 0; r < 8; ++r) rowp[m][r] = 0.0f;

    #pragma unroll
    for (int t = 0; t < 4; ++t) {
        const int   e   = e0 + t * 16 + row;   // C-matrix N = lane&15
        const float hbe = hb[b * DD + e];
        const float ve  = vvec[e];
        #pragma unroll
        for (int m = 0; m < 2; ++m) {
            #pragma unroll
            for (int r = 0; r < 8; ++r) {
                float p = tanhf(acc[m][t][r] + hbe) * ve;
                // reduce over the 16 N-lanes of this row (width=16 keeps halves separate)
                p += __shfl_xor(p, 1, 16);
                p += __shfl_xor(p, 2, 16);
                p += __shfl_xor(p, 4, 16);
                p += __shfl_xor(p, 8, 16);
                rowp[m][r] += p;               // valid on lanes 0 (rows r) and 16 (rows r+8)
            }
        }
    }

    if (row == 0) {
        #pragma unroll
        for (int m = 0; m < 2; ++m)
            #pragma unroll
            for (int r = 0; r < 8; ++r)
                red[wave][m * 16 + r + h8] = rowp[m][r];
    }
    __syncthreads();

    // cross-wave reduce: one atomic per row per block
    if (threadIdx.x < 32) {
        float sum = 0.0f;
        #pragma unroll
        for (int w = 0; w < 8; ++w) sum += red[w][threadIdx.x];
        atomicAdd(&scores[b * SS + s0 + threadIdx.x], sum);
    }
}

// ---------------- kernel 3: masked softmax over S, in place (scores -> attn) ----------------
__global__ __launch_bounds__(256) void k_softmax(float* __restrict__ scores,
                                                 const int* __restrict__ lengths) {
    __shared__ float red[256];
    const int b   = blockIdx.x;
    const int len = lengths[b];
    float* sc = scores + b * SS;

    float m = -INFINITY;
    for (int s = threadIdx.x; s < len; s += 256) m = fmaxf(m, sc[s]);
    red[threadIdx.x] = m; __syncthreads();
    for (int off = 128; off > 0; off >>= 1) {
        if (threadIdx.x < off) red[threadIdx.x] = fmaxf(red[threadIdx.x], red[threadIdx.x + off]);
        __syncthreads();
    }
    m = red[0]; __syncthreads();

    float sum = 0.0f;
    for (int s = threadIdx.x; s < SS; s += 256) {
        float ev = (s < len) ? __expf(sc[s] - m) : 0.0f;
        sc[s] = ev;
        sum += ev;
    }
    red[threadIdx.x] = sum; __syncthreads();
    for (int off = 128; off > 0; off >>= 1) {
        if (threadIdx.x < off) red[threadIdx.x] += red[threadIdx.x + off];
        __syncthreads();
    }
    const float inv = 1.0f / red[0];
    for (int s = threadIdx.x; s < SS; s += 256) sc[s] *= inv;
}

// ---------------- kernel 4: context[b,d] = sum_s attn[b,s] * enc[b,s,d] ----------------
__global__ __launch_bounds__(256) void k_context(const float* __restrict__ enc,
                                                 const float* __restrict__ attn,
                                                 float* __restrict__ out) {
    const int d = blockIdx.x * 256 + threadIdx.x;   // grid.x = DD/256
    const int b = blockIdx.y;
    const float* a  = attn + b * SS;
    const float* ep = enc + (size_t)b * SS * DD + d;
    float acc = 0.0f;
    #pragma unroll 4
    for (int s = 0; s < SS; ++s)
        acc += a[s] * ep[(size_t)s * DD];
    out[b * DD + d] = acc;
}

extern "C" void kernel_launch(void* const* d_in, const int* in_sizes, int n_in,
                              void* d_out, int out_size, void* d_ws, size_t ws_size,
                              hipStream_t stream) {
    const float* enc     = (const float*)d_in[0];  // (B,S,D)
    const float* hidden  = (const float*)d_in[1];  // (B,D)
    const int*   lengths = (const int*)  d_in[2];  // (B,)
    const float* W       = (const float*)d_in[3];  // (D, 2D)
    const float* bias    = (const float*)d_in[4];  // (D,)
    const float* vvec    = (const float*)d_in[5];  // (D,)
    float* out = (float*)d_out;                    // (B,1,D)

    float* hb     = (float*)d_ws;                  // B*D floats
    float* scores = hb + BB * DD;                  // B*S floats (becomes attn in-place)

    k_zero   <<<(BB * SS + 255) / 256, 256, 0, stream>>>(scores, BB * SS);
    k_hidproj<<<dim3(DD / 256, BB), 256, 0, stream>>>(hidden, W, bias, hb);
    k_energy <<<dim3(DD / 512, SS / 32, BB), 256, 0, stream>>>(enc, W, hb, vvec, scores);
    k_softmax<<<BB, 256, 0, stream>>>(scores, lengths);
    k_context<<<dim3(DD / 256, BB), 256, 0, stream>>>(enc, scores, out);
}
